// PolaLinearAttention_52888227283137
// MI455X (gfx1250) — matrix operand order repaired and checked by reading, not compiled, not run on hardware
//
#include <hip/hip_runtime.h>
#include <cmath>

// ---------------- problem constants (from setup_inputs) ----------------
#define Bv   16
#define NN   3136          // N = 56*56
#define Cc   512
#define NHd  8
#define HDd  64
#define BN   (Bv*NN)       // 50176 tokens
#define HH   56

typedef __attribute__((ext_vector_type(8)))  _Float16 h8;
typedef __attribute__((ext_vector_type(16))) _Float16 hv16;
typedef __attribute__((ext_vector_type(8)))  float    f8;
typedef __attribute__((ext_vector_type(4)))  int      i4;

// ---- async global->LDS copy (gfx1250 GLOBAL_LOAD_ASYNC_TO_LDS_B128) ----
#if defined(__has_builtin)
#if __has_builtin(__builtin_amdgcn_global_load_async_to_lds_b128) && \
    __has_builtin(__builtin_amdgcn_s_wait_asynccnt)
#define HAVE_ASYNC_LDS 1
#endif
#endif

#ifdef HAVE_ASYNC_LDS
__device__ __forceinline__ void async_cp16(const _Float16* gsrc, _Float16* ldst) {
  __builtin_amdgcn_global_load_async_to_lds_b128(
      (__attribute__((address_space(1))) i4*)gsrc,
      (__attribute__((address_space(3))) i4*)ldst, 0, 0);
}
__device__ __forceinline__ void async_wait() {
  __builtin_amdgcn_s_wait_asynccnt(0);
}
#else
__device__ __forceinline__ void async_cp16(const _Float16* gsrc, _Float16* ldst) {
  *(h8*)ldst = *(const h8*)gsrc;
}
__device__ __forceinline__ void async_wait() {}
#endif

__device__ __forceinline__ f8 zero_f8() {
  f8 a;
#pragma unroll
  for (int i = 0; i < 8; ++i) a[i] = 0.f;
  return a;
}

__device__ __forceinline__ f8 wmma16(hv16 a, hv16 b, f8 c) {
  // v_wmma_f32_16x16x32_f16  (neg_a, A, neg_b, B, c_mod, C, reuse_a, reuse_b)
  return __builtin_amdgcn_wmma_f32_16x16x32_f16(false, a, false, b, (short)0, c,
                                                false, false);
}

// A-fragment (16x32 f16, M-major): lane<16 -> K 0..7 and 16..23 of row lane;
// lane>=16 -> K 8..15 and 24..31.  p points at [row][ (lane>>4)*8 ].
__device__ __forceinline__ hv16 ld_fragA(const _Float16* p) {
  h8 lo = *(const h8*)p;
  h8 hi = *(const h8*)(p + 16);
  hv16 r;
#pragma unroll
  for (int i = 0; i < 8; ++i) { r[i] = lo[i]; r[i + 8] = hi[i]; }
  return r;
}

// x^p for x>0 via v_log_f32 (log2) + v_exp_f32 (exp2); 0 otherwise.
__device__ __forceinline__ float pow_relu(float x, float p) {
  return (x > 0.f) ? __builtin_amdgcn_exp2f(p * __builtin_amdgcn_logf(x)) : 0.f;
}

// ---------------- tiny prep kernels ----------------
__global__ void prep_params_kernel(const float* __restrict__ scale_p,
                                   const float* __restrict__ power_p,
                                   float* __restrict__ inv_scale,
                                   float* __restrict__ powv) {
  int i = blockIdx.x * blockDim.x + threadIdx.x;
  if (i < Cc) {
    float sp = scale_p[i];
    float softplus = (sp > 20.f) ? sp : log1pf(expf(sp));
    inv_scale[i] = 1.f / softplus;
    powv[i] = 1.f + 4.f / (1.f + expf(-power_p[i]));   // 1 + ALPHA*sigmoid
  }
}

__global__ void f32_to_f16_kernel(const float* __restrict__ in,
                                  _Float16* __restrict__ out, int n) {
  int i = blockIdx.x * blockDim.x + threadIdx.x;
  if (i < n) out[i] = (_Float16)in[i];
}

// weights: (K, Ncols) f32 -> (Ncols, K) f16, one-off tiny transpose
__global__ void f32_to_f16_transpose_kernel(const float* __restrict__ in,
                                            _Float16* __restrict__ out,
                                            int K, int Ncols) {
  int idx = blockIdx.x * blockDim.x + threadIdx.x;
  if (idx < K * Ncols) {
    int n = idx / K, k = idx - n * K;
    out[idx] = (_Float16)in[(size_t)k * Ncols + n];
  }
}

// ---------------- main WMMA GEMM with fused epilogues ----------------
// C[M x Ncols] = A[M x K] (f16) @ Wt[Ncols x K]^T (f16), tile 128x128x32, 8 waves.
// Wt is the PRE-TRANSPOSED weight (row n holds column n of W) so both LDS
// stagings are contiguous 16B async copies.
// mode 0: qg GEMM  -> q features (qf, [B,H,N,128] f16) + gate g (f16)
// mode 1: kv GEMM  -> k features transposed (kfT, [B,H,128,N]) + v planar (vT, [B,H,64,N])
// mode 2: proj GEMM -> out f32 + bias
__global__ __launch_bounds__(256)
void gemm_fused_kernel(const _Float16* __restrict__ A,
                       const _Float16* __restrict__ Wt,
                       int K, int Ncols, int mode,
                       const float* __restrict__ inv_scale,
                       const float* __restrict__ powv,
                       const float* __restrict__ pos_enc,
                       _Float16* __restrict__ featDst,   // qf (0) / kfT (1)
                       _Float16* __restrict__ auxDst,    // g  (0) / vT  (1)
                       const float* __restrict__ proj_b,
                       float* __restrict__ outF) {
  __shared__ __align__(32) _Float16 ldsA[128 * 32];
  __shared__ __align__(32) _Float16 ldsBt[128 * 32];   // stored (n, k)

  const int tid = threadIdx.x;
  const int m0 = blockIdx.x * 128;
  const int n0 = blockIdx.y * 128;
  const int w = tid >> 5, lane = tid & 31;
  const int wm = w >> 2, wn = w & 3;                   // 2 x 4 wave grid
  const int lm = lane & 15;
  const int lk8 = (lane >> 4) * 8, lk16 = (lane >> 4) * 16;

  f8 acc[4][2];
#pragma unroll
  for (int i = 0; i < 4; ++i)
#pragma unroll
    for (int j = 0; j < 2; ++j) acc[i][j] = zero_f8();

  // per-thread staging chunks: c0/c1 of 512 16B chunks covering a 128x32 tile
  const int c0 = tid * 2, c1 = tid * 2 + 1;
  const int arow0 = c0 >> 2, akc0 = (c0 & 3) * 8;
  const int arow1 = c1 >> 2, akc1 = (c1 & 3) * 8;

  for (int k0 = 0; k0 < K; k0 += 32) {
    // stage A tile (128x32 row-major) and Bt tile (128x32, rows = W columns)
    async_cp16(A + (size_t)(m0 + arow0) * K + k0 + akc0, ldsA + arow0 * 32 + akc0);
    async_cp16(A + (size_t)(m0 + arow1) * K + k0 + akc1, ldsA + arow1 * 32 + akc1);
    async_cp16(Wt + (size_t)(n0 + arow0) * K + k0 + akc0, ldsBt + arow0 * 32 + akc0);
    async_cp16(Wt + (size_t)(n0 + arow1) * K + k0 + akc1, ldsBt + arow1 * 32 + akc1);
    if (k0 + 32 < K)  // hint next A tile into cache (global_prefetch_b8)
      __builtin_prefetch(A + (size_t)(m0 + arow0) * K + k0 + 32, 0, 1);
    async_wait();
    __syncthreads();

    hv16 af[4], bfv[2];
#pragma unroll
    for (int fm = 0; fm < 4; ++fm) {
      int row = wm * 64 + fm * 16 + lm;
      af[fm] = ld_fragA(ldsA + row * 32 + lk8);
    }
#pragma unroll
    for (int fn = 0; fn < 2; ++fn) {
      int col = wn * 32 + fn * 16 + lm;
      bfv[fn] = ld_fragA(ldsBt + col * 32 + lk8);  // K lk8..+7 and 16+lk8..+7
    }
#pragma unroll
    for (int fm = 0; fm < 4; ++fm)
#pragma unroll
      for (int fn = 0; fn < 2; ++fn)
        acc[fm][fn] = wmma16(af[fm], bfv[fn], acc[fm][fn]);
    __syncthreads();
  }

  // -------- fused epilogue --------
#pragma unroll
  for (int fm = 0; fm < 4; ++fm)
#pragma unroll
    for (int fn = 0; fn < 2; ++fn)
#pragma unroll
      for (int r = 0; r < 8; ++r) {
        int m_g = m0 + wm * 64 + fm * 16 + lk8 + r;     // M = r + 8*(lane>=16)
        int col = n0 + wn * 32 + fn * 16 + lm;
        float val = acc[fm][fn][r];
        if (mode == 2) {
          outF[(size_t)m_g * Ncols + col] = val + proj_b[col];
        } else {
          int b = m_g / NN, nn = m_g - b * NN;
          if (col < Cc) {
            if (mode == 1) val += pos_enc[nn * Cc + col];  // k += pos_enc
            float c = val * inv_scale[col];                // q/scale, k/scale
            float p = powv[col];
            float fpos = pow_relu(c, p);
            float fneg = pow_relu(-c, p);
            int h = col >> 6, d = col & 63;
            if (mode == 0) {       // qf: [b,h,n, d | 64+d]
              _Float16* dst = featDst + (((size_t)(b * NHd + h)) * NN + nn) * 128;
              dst[d] = (_Float16)fpos;
              dst[64 + d] = (_Float16)fneg;
            } else {               // kfT: [b,h, d | 64+d, n]
              _Float16* dst = featDst + ((size_t)(b * NHd + h) * 128 + d) * NN + nn;
              dst[0] = (_Float16)fpos;
              dst[(size_t)64 * NN] = (_Float16)fneg;
            }
          } else {
            int e = col - Cc;
            if (mode == 0) {       // gate g, token-major f16
              auxDst[(size_t)m_g * Cc + e] = (_Float16)val;
            } else {               // vT: [b,h, e, n] channel-planar
              int h = e >> 6, ee = e & 63;
              auxDst[((size_t)(b * NHd + h) * 64 + ee) * NN + nn] = (_Float16)val;
            }
          }
        }
      }
}

// ---------------- km = mean_n kf  (one block per (b,h,d128) row) ----------------
__global__ __launch_bounds__(256)
void km_kernel(const _Float16* __restrict__ kfT, float* __restrict__ km) {
  __shared__ float red[256];
  int row = blockIdx.x;                       // b*NH*128 rows
  const _Float16* p = kfT + (size_t)row * NN;
  float s = 0.f;
  for (int i = threadIdx.x; i < NN; i += 256) s += (float)p[i];
  red[threadIdx.x] = s;
  __syncthreads();
  for (int st = 128; st > 0; st >>= 1) {
    if (threadIdx.x < st) red[threadIdx.x] += red[threadIdx.x + st];
    __syncthreads();
  }
  if (threadIdx.x == 0) km[row] = red[0] * (1.f / NN);
}

// ---------------- z denominators: z_sim = 1/(qf.km+eps), z_opp uses swapped km --
__global__ void z_kernel(const _Float16* __restrict__ qf,
                         const float* __restrict__ km,
                         float* __restrict__ zsim, float* __restrict__ zopp) {
  int idx = blockIdx.x * blockDim.x + threadIdx.x;
  if (idx >= Bv * NHd * NN) return;
  int bh = idx / NN, n = idx - bh * NN;
  const _Float16* q = qf + ((size_t)bh * NN + n) * 128;
  const float* kmr = km + bh * 128;
  float s1 = 0.f, s2 = 0.f;
#pragma unroll 8
  for (int d = 0; d < 128; ++d) {
    float qv = (float)q[d];
    s1 += qv * kmr[d];
    s2 += qv * kmr[(d + 64) & 127];   // q_opp . km == q_sim . swap(km)
  }
  zsim[idx] = 1.f / (s1 + 1e-6f);
  zopp[idx] = 1.f / (s2 + 1e-6f);
}

// ---------------- kv_ GEMM: per (b,h) 128x64 = kfT(128xN) @ v(Nx64), K=3136 ----
// Output stored transposed (e,d) with opp-branch rows pre-swapped so the next
// GEMM is a single qf @ B.
__global__ __launch_bounds__(256)
void kvmat_kernel(const _Float16* __restrict__ kfT,
                  const _Float16* __restrict__ vT,
                  _Float16* __restrict__ kvT) {
  const int bh = blockIdx.x;
  const _Float16* Ab = kfT + (size_t)bh * 128 * NN;
  const _Float16* Bb = vT + (size_t)bh * 64 * NN;
  const int tid = threadIdx.x, w = tid >> 5, lane = tid & 31;
  const int wm = w >> 2, wn = w & 3;              // 2 x 4 -> 64(d) x 16(e)
  const int lm = lane & 15;
  const int lk8 = (lane >> 4) * 8, lk16 = (lane >> 4) * 16;

  f8 acc[4];
#pragma unroll
  for (int i = 0; i < 4; ++i) acc[i] = zero_f8();

  const _Float16* bp = Bb + (size_t)(wn * 16 + lm) * NN + lk8;
  for (int k0 = 0; k0 < NN; k0 += 32) {           // 98 WMMA steps
    hv16 bfr = ld_fragA(bp + k0);           // B col = v channel plane
#pragma unroll
    for (int fm = 0; fm < 4; ++fm) {
      int row = wm * 64 + fm * 16 + lm;
      hv16 a = ld_fragA(Ab + (size_t)row * NN + k0 + lk8);
      acc[fm] = wmma16(a, bfr, acc[fm]);
    }
  }
  const float inv_n = 1.f / NN;
#pragma unroll
  for (int fm = 0; fm < 4; ++fm)
#pragma unroll
    for (int r = 0; r < 8; ++r) {
      int d_g = wm * 64 + fm * 16 + lk8 + r;
      int e_g = wn * 16 + lm;
      float val = acc[fm][r] * inv_n;
      int dd = (e_g < 32) ? d_g : ((d_g + 64) & 127);  // pre-swap opp rows
      kvT[(size_t)bh * 64 * 128 + (size_t)e_g * 128 + dd] = (_Float16)val;
    }
}

// ---------------- xo GEMM: per (b,h) (Nx128)@(128x64), z-scaled epilogue -------
__global__ __launch_bounds__(256)
void xo_kernel(const _Float16* __restrict__ qf,
               const _Float16* __restrict__ kvT,
               const float* __restrict__ zsim, const float* __restrict__ zopp,
               float* __restrict__ xo) {
  const int bh = blockIdx.y;
  const int m0 = blockIdx.x * 64;
  const _Float16* Ab = qf + (size_t)bh * NN * 128;
  const _Float16* Bb = kvT + (size_t)bh * 64 * 128;
  const int tid = threadIdx.x, w = tid >> 5, lane = tid & 31;
  const int wm = w >> 1, wn = w & 1;              // 4 x 2 -> 16(n) x 32(e)
  const int lm = lane & 15;
  const int lk8 = (lane >> 4) * 8, lk16 = (lane >> 4) * 16;

  f8 acc[2];
  acc[0] = zero_f8(); acc[1] = zero_f8();

  const int row = m0 + wm * 16 + lm;
#pragma unroll
  for (int kt = 0; kt < 4; ++kt) {
    int k0 = kt * 32;
    hv16 a = ld_fragA(Ab + (size_t)row * 128 + k0 + lk8);
#pragma unroll
    for (int fn = 0; fn < 2; ++fn) {
      int col = wn * 32 + fn * 16 + lm;
      hv16 b = ld_fragA(Bb + (size_t)col * 128 + k0 + lk8);
      acc[fn] = wmma16(a, b, acc[fn]);
    }
  }
  const int b = bh >> 3, h = bh & 7;
#pragma unroll
  for (int fn = 0; fn < 2; ++fn)
#pragma unroll
    for (int r = 0; r < 8; ++r) {
      int n_g = m0 + wm * 16 + lk8 + r;
      int e_g = wn * 32 + fn * 16 + lm;
      float z = (e_g < 32 ? zsim : zopp)[(size_t)bh * NN + n_g];
      xo[((size_t)b * NN + n_g) * Cc + h * 64 + e_g] = acc[fn][r] * z;
    }
}

// ---------------- depthwise 5x5 conv on v + combine: tmp = (xo+vd)*g ----------
__global__ __launch_bounds__(256)
void conv_combine_kernel(const _Float16* __restrict__ vT,
                         const float* __restrict__ dwc_w,
                         const float* __restrict__ dwc_b,
                         const float* __restrict__ xo,
                         const _Float16* __restrict__ g,
                         _Float16* __restrict__ tmp) {
  __shared__ __align__(16) _Float16 plane[NN];   // one 56x56 channel plane
  __shared__ float wsh[25];
  const int e = blockIdx.x;          // 0..63 channel in head
  const int bh = blockIdx.y;         // 0..127
  const int tid = threadIdx.x;

  const _Float16* src = vT + ((size_t)bh * 64 + e) * NN;
  for (int c = tid; c < NN / 8; c += 256)        // 392 x 16B async chunks
    async_cp16(src + c * 8, plane + c * 8);
  if (tid < 25) wsh[tid] = dwc_w[e * 25 + tid];
  async_wait();
  __syncthreads();

  const float bias = dwc_b[e];
  const int b = bh >> 3, h = bh & 7;
  for (int n = tid; n < NN; n += 256) {
    int hh = n / HH, ww = n - hh * HH;
    float acc = bias;
#pragma unroll
    for (int dy = 0; dy < 5; ++dy) {
      int y = hh + dy - 2;
      if ((unsigned)y < HH) {
#pragma unroll
        for (int dx = 0; dx < 5; ++dx) {
          int x = ww + dx - 2;
          if ((unsigned)x < HH)
            acc += (float)plane[y * HH + x] * wsh[dy * 5 + dx];
        }
      }
    }
    size_t oi = ((size_t)b * NN + n) * Cc + h * 64 + e;
    tmp[oi] = (_Float16)((xo[oi] + acc) * (float)g[oi]);
  }
}

// ---------------- host launch ----------------
extern "C" void kernel_launch(void* const* d_in, const int* in_sizes, int n_in,
                              void* d_out, int out_size, void* d_ws, size_t ws_size,
                              hipStream_t stream) {
  const float* x       = (const float*)d_in[0];
  const float* qg_w    = (const float*)d_in[1];
  const float* kv_w    = (const float*)d_in[2];
  const float* proj_w  = (const float*)d_in[3];
  const float* proj_b  = (const float*)d_in[4];
  const float* pos_enc = (const float*)d_in[5];
  const float* scale_p = (const float*)d_in[6];
  const float* power_p = (const float*)d_in[7];
  const float* dwc_w   = (const float*)d_in[8];
  const float* dwc_b   = (const float*)d_in[9];
  float* out = (float*)d_out;

  // workspace carve-out (256B aligned)
  char* ws = (char*)d_ws;
  size_t off = 0;
  auto take = [&](size_t bytes) -> char* {
    char* p = ws + off;
    off += (bytes + 255) & ~(size_t)255;
    return p;
  };
  _Float16* xh   = (_Float16*)take((size_t)BN * Cc * 2);        // later reused as tmp
  _Float16* wqgT = (_Float16*)take((size_t)Cc * 2 * Cc * 2);    // (1024, 512) f16
  _Float16* wkvT = (_Float16*)take((size_t)Cc * 2 * Cc * 2);
  _Float16* wpjT = (_Float16*)take((size_t)Cc * Cc * 2);        // (512, 512) f16
  float* inv_scale = (float*)take(Cc * 4);
  float* powv      = (float*)take(Cc * 4);
  _Float16* qf   = (_Float16*)take((size_t)Bv * NHd * NN * 128 * 2);
  _Float16* kfT  = (_Float16*)take((size_t)Bv * NHd * NN * 128 * 2); // reused as xo f32
  _Float16* gh   = (_Float16*)take((size_t)BN * Cc * 2);
  _Float16* vT   = (_Float16*)take((size_t)Bv * NHd * 64 * NN * 2);
  float* km      = (float*)take((size_t)Bv * NHd * 128 * 4);
  _Float16* kvT  = (_Float16*)take((size_t)Bv * NHd * 64 * 128 * 2);
  float* zsim    = (float*)take((size_t)Bv * NHd * NN * 4);
  float* zopp    = (float*)take((size_t)Bv * NHd * NN * 4);
  float* xo      = (float*)kfT;    // alias: kfT dead after kvmat/km
  _Float16* tmph = xh;             // alias: x dead after the two input GEMMs

  // 0) parameters, x conversion, weight transposed conversions
  prep_params_kernel<<<2, 256, 0, stream>>>(scale_p, power_p, inv_scale, powv);
  {
    int n = BN * Cc;
    f32_to_f16_kernel<<<(n + 255) / 256, 256, 0, stream>>>(x, xh, n);
    n = Cc * 2 * Cc;
    f32_to_f16_transpose_kernel<<<(n + 255) / 256, 256, 0, stream>>>(qg_w, wqgT, Cc, 2 * Cc);
    f32_to_f16_transpose_kernel<<<(n + 255) / 256, 256, 0, stream>>>(kv_w, wkvT, Cc, 2 * Cc);
    n = Cc * Cc;
    f32_to_f16_transpose_kernel<<<(n + 255) / 256, 256, 0, stream>>>(proj_w, wpjT, Cc, Cc);
  }

  // 1) qg GEMM -> qf, g     2) kv GEMM -> kfT, vT
  gemm_fused_kernel<<<dim3(BN / 128, (2 * Cc) / 128), 256, 0, stream>>>(
      xh, wqgT, Cc, 2 * Cc, 0, inv_scale, powv, pos_enc, qf, gh, nullptr, nullptr);
  gemm_fused_kernel<<<dim3(BN / 128, (2 * Cc) / 128), 256, 0, stream>>>(
      xh, wkvT, Cc, 2 * Cc, 1, inv_scale, powv, pos_enc, kfT, vT, nullptr, nullptr);

  // 3) km, z
  km_kernel<<<Bv * NHd * 128, 256, 0, stream>>>(kfT, km);
  z_kernel<<<(Bv * NHd * NN + 255) / 256, 256, 0, stream>>>(qf, km, zsim, zopp);

  // 4) kv_ = kfT @ v (per head), pre-swapped + transposed for next GEMM
  kvmat_kernel<<<Bv * NHd, 256, 0, stream>>>(kfT, vT, kvT);

  // 5) xo = (qf @ kv_) * z        (xo aliases kfT storage)
  xo_kernel<<<dim3(NN / 64, Bv * NHd), 256, 0, stream>>>(qf, kvT, zsim, zopp, xo);

  // 6) depthwise conv + gate: tmp = (xo + vd) * g   (tmp aliases xh)
  conv_combine_kernel<<<dim3(64, Bv * NHd), 256, 0, stream>>>(vT, dwc_w, dwc_b, xo,
                                                              gh, tmph);

  // 7) out = tmp @ proj_w + proj_b
  gemm_fused_kernel<<<dim3(BN / 128, Cc / 128), 256, 0, stream>>>(
      tmph, wpjT, Cc, Cc, 2, nullptr, nullptr, nullptr, nullptr, nullptr, proj_b, out);
}
